// MSAAttention_19138374271388
// MI455X (gfx1250) — matrix operand order repaired and checked by reading, not compiled, not run on hardware
//
#include <hip/hip_runtime.h>

typedef __attribute__((ext_vector_type(16))) _Float16 v16h;
typedef __attribute__((ext_vector_type(8)))  _Float16 v8h;
typedef __attribute__((ext_vector_type(8)))  float    v8f;

constexpr int HD  = 512;   // hidden
constexpr int SEQ = 2048;  // sequence
constexpr int NB  = 8;     // batch

// ---------------- helpers ----------------

__device__ inline v8f zero8() {
    v8f z;
#pragma unroll
    for (int i = 0; i < 8; ++i) z[i] = 0.0f;
    return z;
}

// A-operand fragment (16x32 f16): lane L (row = L&15) holds 16 contiguous
// K-halves starting at kgroup (=16 for L>=16).
__device__ inline v16h make_a(const _Float16* p) {
    v8h lo = *(const v8h*)p;
    v8h hi = *(const v8h*)(p + 8);
    v16h r;
#pragma unroll
    for (int i = 0; i < 8; ++i) { r[i] = lo[i]; r[i + 8] = hi[i]; }
    return r;
}

// B-operand fragment (32x16 f16): lane L (col = L&15) holds 16 contiguous
// K-halves starting at kgroup (=16 for L>=16).
__device__ inline v16h make_b(const _Float16* p) {
    v8h lo = *(const v8h*)p;
    v8h hi = *(const v8h*)(p + 8);
    v16h r;
#pragma unroll
    for (int i = 0; i < 8; ++i) { r[i] = lo[i]; r[i + 8] = hi[i]; }
    return r;
}

__device__ inline v8f wmma_f16(v16h a, v16h b, v8f c) {
    return __builtin_amdgcn_wmma_f32_16x16x32_f16(false, a, false, b, (short)0, c,
                                                  false, false);
}

// Stage a 16KB tile (16 rows x 512 f16) global -> LDS using CDNA5 async
// LDS-direct loads (ASYNCcnt-tracked, no VGPR round trip). 256 threads move
// 4 x 16B chunks each; INST_OFFSET applies to both LDS and global addresses
// (ISA 08_async_tensor §4.4), so one address pair covers all four.
__device__ inline void async_stage_16k(const _Float16* __restrict__ g,
                                       _Float16* __restrict__ l, int tid) {
    unsigned           lds = (unsigned)(size_t)l + (unsigned)tid * 16u;
    unsigned long long ga  = (unsigned long long)(size_t)g +
                             (unsigned long long)tid * 16ull;
    asm volatile(
        "global_load_async_to_lds_b128 %0, %1, off\n\t"
        "global_load_async_to_lds_b128 %0, %1, off offset:4096\n\t"
        "global_load_async_to_lds_b128 %0, %1, off offset:8192\n\t"
        "global_load_async_to_lds_b128 %0, %1, off offset:12288"
        :
        : "v"(lds), "v"(ga)
        : "memory");
    asm volatile("s_wait_asynccnt 0x0" ::: "memory");
}

// ---------------- prep kernels ----------------

__global__ void k_pe_cast(const float* __restrict__ x, _Float16* __restrict__ xh) {
    int idx = blockIdx.x * blockDim.x + threadIdx.x;       // [0, NB*SEQ*HD)
    int h = idx & (HD - 1);
    int s = (idx >> 9) & (SEQ - 1);
    int i = h & 255;
    float ang = (float)s * __expf((float)(2 * i) * (-9.210340371976184f / 512.0f));
    float pe = (h < 256) ? __sinf(ang) : __cosf(ang);
    xh[idx] = (_Float16)(x[idx] + pe);
}

__global__ void k_cast(const float* __restrict__ src, _Float16* __restrict__ dst, int n) {
    int i = blockIdx.x * blockDim.x + threadIdx.x;
    if (i < n) dst[i] = (_Float16)src[i];
}

// ---------------- QKV projection: Y = Xf16 @ W^T + b ----------------
__global__ __launch_bounds__(256) void k_qkv(const _Float16* __restrict__ Xh,
                                             const _Float16* __restrict__ Wh,
                                             const float* __restrict__ bias,
                                             _Float16* __restrict__ Y,
                                             _Float16* __restrict__ YT) {
    __shared__ __align__(16) _Float16 Xs[16][HD];

    int tid = threadIdx.x, w = tid >> 5, lane = tid & 31;
    int row0 = blockIdx.x * 16;

    async_stage_16k(Xh + (size_t)row0 * HD, &Xs[0][0], tid);
    __syncthreads();

    int arow  = lane & 15;
    int ab    = (lane >> 4) << 4;   // A k-offset: 0 / 16
    int bb    = (lane >> 4) << 4;   // B k-offset: 0 / 16
    int half8 = (lane >> 4) << 3;   // C row base: 0 / 8

    v8f acc[4];
#pragma unroll
    for (int t = 0; t < 4; ++t) acc[t] = zero8();

#pragma unroll
    for (int kd = 0; kd < 16; ++kd) {
        v16h A = make_a(&Xs[arow][kd * 32 + ab]);
#pragma unroll
        for (int t = 0; t < 4; ++t) {
            int d = w * 64 + t * 16 + arow;
            v16h Bf = make_b(Wh + (size_t)d * HD + kd * 32 + bb);
            acc[t] = wmma_f16(A, Bf, acc[t]);
        }
    }

    if (YT == nullptr) {
#pragma unroll
        for (int t = 0; t < 4; ++t) {
            int d = w * 64 + t * 16 + arow;
            float bv = bias[d];
#pragma unroll
            for (int r = 0; r < 8; ++r)
                Y[(size_t)(row0 + half8 + r) * HD + d] = (_Float16)(acc[t][r] + bv);
        }
    } else {
        int b  = row0 >> 11;
        int s0 = row0 & (SEQ - 1);
#pragma unroll
        for (int t = 0; t < 4; ++t) {
            int d = w * 64 + t * 16 + arow;
            float bv = bias[d];
            v8h pk;
#pragma unroll
            for (int r = 0; r < 8; ++r) pk[r] = (_Float16)(acc[t][r] + bv);
            *(v8h*)(YT + ((size_t)(b * HD + d)) * SEQ + s0 + half8) = pk;
        }
    }
}

// ---------------- flash attention ----------------
__global__ __launch_bounds__(256) void k_flash(const _Float16* __restrict__ Qh,
                                               const _Float16* __restrict__ Kh,
                                               const _Float16* __restrict__ VTh,
                                               float* __restrict__ out) {
    __shared__ __align__(16) _Float16 Qs[16][HD];
    __shared__ __align__(16) _Float16 Ps[16][128];
    __shared__ float redm[8][16];
    __shared__ float reds[8][16];

    int tid = threadIdx.x, w = tid >> 5, lane = tid & 31;
    int b = blockIdx.y, q0 = blockIdx.x * 16;

    async_stage_16k(Qh + ((size_t)b * SEQ + q0) * HD, &Qs[0][0], tid);
    __syncthreads();

    int arow  = lane & 15;
    int ab    = (lane >> 4) << 4;
    int bb    = (lane >> 4) << 4;
    int half8 = (lane >> 4) << 3;

    float mrun[8], lrun[8];
    v8f Oacc[4];
#pragma unroll
    for (int r = 0; r < 8; ++r) { mrun[r] = -3.0e38f; lrun[r] = 0.0f; }
#pragma unroll
    for (int t = 0; t < 4; ++t) Oacc[t] = zero8();

    for (int kt = 0; kt < SEQ; kt += 128) {
        const _Float16* Kg = Kh + ((size_t)b * SEQ + kt + w * 16 + arow) * HD + bb;

        // prefetch next KV tile toward L2 while this tile computes
        if (kt + 128 < SEQ) {
            __builtin_prefetch(Kg + (size_t)128 * HD, 0, 1);
            __builtin_prefetch(VTh + ((size_t)b * HD + w * 64 + arow) * SEQ +
                                   kt + 128 + bb, 0, 1);
        }

        // ---- scores S = Q @ K^T for this wave's 16 keys
        v8f Sc = zero8();
#pragma unroll
        for (int kd = 0; kd < 16; ++kd) {
            v16h A  = make_a(&Qs[arow][kd * 32 + ab]);
            v16h Bf = make_b(Kg + kd * 32);
            Sc = wmma_f16(A, Bf, Sc);
        }

        // ---- row max across the 16 lanes of this half-wave
        float rmx[8];
#pragma unroll
        for (int r = 0; r < 8; ++r) rmx[r] = Sc[r];
#pragma unroll
        for (int off = 8; off; off >>= 1)
#pragma unroll
            for (int r = 0; r < 8; ++r)
                rmx[r] = fmaxf(rmx[r], __shfl_xor(rmx[r], off, 32));
        if (arow == 0)
#pragma unroll
            for (int r = 0; r < 8; ++r) redm[w][half8 + r] = rmx[r];
        __syncthreads();

        // ---- global row max, rescale factor
        float mnew[8], scale[8];
#pragma unroll
        for (int r = 0; r < 8; ++r) {
            float m = mrun[r];
#pragma unroll
            for (int ww = 0; ww < 8; ++ww) m = fmaxf(m, redm[ww][half8 + r]);
            mnew[r]  = m;
            scale[r] = __expf(mrun[r] - m);
            mrun[r]  = m;
        }

        // ---- P = exp(S - m), row sums, stash P(f16) into LDS
        float p[8], rs[8];
#pragma unroll
        for (int r = 0; r < 8; ++r) { p[r] = __expf(Sc[r] - mnew[r]); rs[r] = p[r]; }
#pragma unroll
        for (int off = 8; off; off >>= 1)
#pragma unroll
            for (int r = 0; r < 8; ++r) rs[r] += __shfl_xor(rs[r], off, 32);
        if (arow == 0)
#pragma unroll
            for (int r = 0; r < 8; ++r) reds[w][half8 + r] = rs[r];
#pragma unroll
        for (int r = 0; r < 8; ++r)
            Ps[half8 + r][w * 16 + arow] = (_Float16)p[r];
        __syncthreads();

        // ---- update l, rescale O
#pragma unroll
        for (int r = 0; r < 8; ++r) {
            float tot = 0.0f;
#pragma unroll
            for (int ww = 0; ww < 8; ++ww) tot += reds[ww][half8 + r];
            lrun[r] = lrun[r] * scale[r] + tot;
        }
#pragma unroll
        for (int t = 0; t < 4; ++t)
#pragma unroll
            for (int r = 0; r < 8; ++r) Oacc[t][r] *= scale[r];

        // ---- O += P @ V (VT is pre-transposed: contiguous per-lane B loads)
#pragma unroll
        for (int t = 0; t < 4; ++t) {
            int d = w * 64 + t * 16 + arow;
            const _Float16* Vg = VTh + ((size_t)b * HD + d) * SEQ + kt + bb;
#pragma unroll
            for (int ks = 0; ks < 4; ++ks) {
                v16h A  = make_a(&Ps[arow][ks * 32 + ab]);
                v16h Bf = make_b(Vg + ks * 32);
                Oacc[t] = wmma_f16(A, Bf, Oacc[t]);
            }
        }
        __syncthreads();   // protect Ps/redm/reds for next tile
    }

    // ---- normalize and store fp32 output
#pragma unroll
    for (int t = 0; t < 4; ++t)
#pragma unroll
        for (int r = 0; r < 8; ++r)
            out[((size_t)b * SEQ + q0 + half8 + r) * HD + w * 64 + t * 16 + arow] =
                Oacc[t][r] / lrun[r];
}

// ---------------- launch ----------------

extern "C" void kernel_launch(void* const* d_in, const int* in_sizes, int n_in,
                              void* d_out, int out_size, void* d_ws, size_t ws_size,
                              hipStream_t stream) {
    (void)in_sizes; (void)n_in; (void)out_size; (void)ws_size;
    const float* x  = (const float*)d_in[0];
    const float* Wq = (const float*)d_in[1];
    const float* bq = (const float*)d_in[2];
    const float* Wk = (const float*)d_in[3];
    const float* bk = (const float*)d_in[4];
    const float* Wv = (const float*)d_in[5];
    const float* bv = (const float*)d_in[6];
    float* out = (float*)d_out;

    char* ws = (char*)d_ws;
    const size_t MB = 1024 * 1024;
    _Float16* Xh  = (_Float16*)(ws);             // 16 MB : x + PE (f16)
    _Float16* Qh  = (_Float16*)(ws + 16 * MB);   // 16 MB
    _Float16* Kh  = (_Float16*)(ws + 32 * MB);   // 16 MB
    _Float16* VTh = (_Float16*)(ws + 48 * MB);   // 16 MB : V transposed [B][H][S]
    _Float16* Wqh = (_Float16*)(ws + 64 * MB);   // 0.5 MB each
    _Float16* Wkh = Wqh + 512 * 512;
    _Float16* Wvh = Wkh + 512 * 512;

    k_pe_cast<<<NB * SEQ * HD / 256, 256, 0, stream>>>(x, Xh);
    k_cast<<<512 * 512 / 256, 256, 0, stream>>>(Wq, Wqh, 512 * 512);
    k_cast<<<512 * 512 / 256, 256, 0, stream>>>(Wk, Wkh, 512 * 512);
    k_cast<<<512 * 512 / 256, 256, 0, stream>>>(Wv, Wvh, 512 * 512);

    k_qkv<<<NB * SEQ / 16, 256, 0, stream>>>(Xh, Wqh, bq, Qh, nullptr);
    k_qkv<<<NB * SEQ / 16, 256, 0, stream>>>(Xh, Wkh, bk, Kh, nullptr);
    k_qkv<<<NB * SEQ / 16, 256, 0, stream>>>(Xh, Wvh, bv, nullptr, VTh);

    k_flash<<<dim3(SEQ / 16, NB), 256, 0, stream>>>(Qh, Kh, VTh, out);
}
